// CrossAttention_2113123910086
// MI455X (gfx1250) — matrix operand order repaired and checked by reading, not compiled, not run on hardware
//
#include <hip/hip_runtime.h>

#define DEV __device__ __forceinline__

typedef __attribute__((ext_vector_type(8)))  __bf16       v8bf;
typedef __attribute__((ext_vector_type(16))) __bf16       v16bf;
typedef __attribute__((ext_vector_type(8)))  float        v8f;
typedef __attribute__((ext_vector_type(4)))  float        f4;
typedef __attribute__((ext_vector_type(4)))  unsigned int u4;
typedef __attribute__((ext_vector_type(4)))  int          i4;

constexpr int Bb = 4, T = 2048, C = 1024, H = 16, Dh = 64;
constexpr int M = Bb * T;   // 8192
constexpr int K = C;        // 1024
constexpr int N = C;        // 1024

#if __has_builtin(__builtin_amdgcn_global_load_async_to_lds_b128) && \
    __has_builtin(__builtin_amdgcn_s_wait_asynccnt)
#define HAS_ASYNC_LDS 1
#else
#define HAS_ASYNC_LDS 0
#endif

// ---------- helpers ----------
DEV unsigned short f2bf(float f) {            // f32 -> bf16 (RNE)
  unsigned u = __builtin_bit_cast(unsigned, f);
  u += 0x7FFFu + ((u >> 16) & 1u);
  return (unsigned short)(u >> 16);
}
DEV __bf16 f2bfh(float f) {
  unsigned short s = f2bf(f);
  return __builtin_bit_cast(__bf16, s);
}
DEV v8bf cvt8(const float* p) {               // 8 contiguous f32 -> v8bf
  f4 a = *(const f4*)p;
  f4 b = *(const f4*)(p + 4);
  v8bf r;
#pragma unroll
  for (int i = 0; i < 4; ++i) { r[i] = f2bfh(a[i]); r[4 + i] = f2bfh(b[i]); }
  return r;
}
DEV v16bf cat16(v8bf a, v8bf b) {
  return __builtin_shufflevector(a, b, 0,1,2,3,4,5,6,7,8,9,10,11,12,13,14,15);
}
DEV v8f wmma_bf16(v16bf a, v16bf b, v8f c) {
  // D = A(16x32 bf16) * B(32x16 bf16) + C(f32)
  return __builtin_amdgcn_wmma_f32_16x16x32_bf16(false, a, false, b, (short)0, c,
                                                 false, false);
}

// A-fragment (16x32, row M = lane&15), contiguous k order:
// lanes 0-15: elems 0..15 = K k0+0..15
// lanes 16-31: elems 0..15 = K k0+16..31
DEV v16bf load_a_bf(const __bf16* base, int row, int stride, int k0, int hi) {
  const __bf16* p = base + (size_t)row * stride + k0 + (hi ? 16 : 0);
  v8bf lo = *(const v8bf*)p;
  v8bf up = *(const v8bf*)(p + 8);
  return cat16(lo, up);
}
DEV v16bf load_a_f32(const float* base, int row, int stride, int k0, int hi) {
  const float* p = base + (size_t)row * stride + k0 + (hi ? 16 : 0);
  return cat16(cvt8(p), cvt8(p + 8));
}
// B-fragment (32x16, col N = lane&15): lanes 0-15 hold K k0+0..15,
// lanes 16-31 hold K k0+16..31 -> 16 contiguous elems per lane from
// transposed storage [N][K].
DEV v16bf load_b(const __bf16* base, int rowN, int stride, int k0, int hi) {
  const __bf16* p = base + (size_t)rowN * stride + k0 + (hi ? 16 : 0);
  v8bf lo = *(const v8bf*)p;
  v8bf up = *(const v8bf*)(p + 8);
  return cat16(lo, up);
}

// one 16-byte lane-copy global -> LDS (async DMA path when available)
DEV void copy16(const __bf16* g, __bf16* l) {
#if HAS_ASYNC_LDS
  __builtin_amdgcn_global_load_async_to_lds_b128(
      (__attribute__((address_space(1))) i4*)(void*)g,
      (__attribute__((address_space(3))) i4*)(void*)l, 0, 0);
#else
  *(v8bf*)l = *(const v8bf*)g;
#endif
}
DEV void async_wait_all() {
#if HAS_ASYNC_LDS
  __builtin_amdgcn_s_wait_asynccnt(0);
#endif
}

// ---------- weight transpose+convert: W[K][N] f32 -> Wt[N][K] bf16 ----------
__global__ __launch_bounds__(256) void transpose_w(const float* __restrict__ W,
                                                   __bf16* __restrict__ Wt) {
  __shared__ float tile[32][33];
  int tx = threadIdx.x & 31, ty = threadIdx.x >> 5;
  int nb = blockIdx.x * 32, kb = blockIdx.y * 32;
#pragma unroll
  for (int i = 0; i < 32; i += 8)
    tile[ty + i][tx] = W[(size_t)(kb + ty + i) * N + nb + tx];
  __syncthreads();
#pragma unroll
  for (int i = 0; i < 32; i += 8)
    ((unsigned short*)Wt)[(size_t)(nb + ty + i) * K + kb + tx] =
        f2bf(tile[tx][ty + i]);
}

// ---------- GEMM: Y[M,N] = A[M,K] * W + bias ----------
// Weight tile (128N x 32K bf16) is staged per-block into double-buffered LDS
// via async DMA; all 8 waves read B-fragments from LDS (4x traffic cut).
// MODE 0: bf16 out [B,H,T,D], scaled (Q gets log2e/sqrt(D) folded in)
// MODE 1: bf16 out [B,H,D,T]  (V transposed -> contiguous stores!)
// MODE 2: f32 out row-major [M,N]
constexpr int BST = 40;   // LDS weight-tile row stride (32 + 8 pad)

template <bool A_F32, int MODE>
__global__ __launch_bounds__(256) void gemm_kernel(const void* __restrict__ Aptr,
                                                   const __bf16* __restrict__ Wt,
                                                   const float* __restrict__ bias,
                                                   void* __restrict__ outp,
                                                   float scale) {
  __shared__ alignas(16) __bf16 wtile[2][128 * BST];

  const int tid  = threadIdx.x;
  const int lane = tid & 31, ln = lane & 15, hi = lane >> 4;
  const int w  = tid >> 5;
  const int wm = w & 3, wn = w >> 2;
  const int Nblk = blockIdx.x * 128;
  const int M0 = blockIdx.y * 128 + wm * 32;
  const int N0 = Nblk + wn * 64;

  // each thread stages two 16B chunks of the 128x32 weight tile
  auto stageB = [&](int k0, int buf) {
#pragma unroll
    for (int c = 0; c < 2; ++c) {
      const int cid = tid * 2 + c;
      const int row = cid >> 2, col = (cid & 3) * 8;
      copy16(Wt + (size_t)(Nblk + row) * K + k0 + col,
             &wtile[buf][row * BST + col]);
    }
  };

  const v8f vzero = {0.f, 0.f, 0.f, 0.f, 0.f, 0.f, 0.f, 0.f};
  v8f acc[2][4];
#pragma unroll
  for (int mt = 0; mt < 2; ++mt)
#pragma unroll
    for (int nt = 0; nt < 4; ++nt) acc[mt][nt] = vzero;

  stageB(0, 0);
  async_wait_all();
  __syncthreads();

#pragma unroll 2
  for (int kt = 0; kt < K / 32; ++kt) {
    const int k0 = kt * 32;
    const int cur = kt & 1, nxt = cur ^ 1;
    if (kt + 1 < K / 32) stageB(k0 + 32, nxt);   // DMA next weight tile

    v16bf a[2], b[4];
#pragma unroll
    for (int mt = 0; mt < 2; ++mt)
      a[mt] = A_F32 ? load_a_f32((const float*)Aptr, M0 + mt * 16 + ln, K, k0, hi)
                    : load_a_bf((const __bf16*)Aptr, M0 + mt * 16 + ln, K, k0, hi);
#pragma unroll
    for (int nt = 0; nt < 4; ++nt)
      b[nt] = load_b(&wtile[cur][0], wn * 64 + nt * 16 + ln, BST, 0, hi);
#pragma unroll
    for (int mt = 0; mt < 2; ++mt)
#pragma unroll
      for (int nt = 0; nt < 4; ++nt)
        acc[mt][nt] = wmma_bf16(a[mt], b[nt], acc[mt][nt]);

    async_wait_all();
    __syncthreads();
  }

  // epilogue: C-layout -> lane holds fixed column n, rows M0+mt*16+hi*8+r
#pragma unroll
  for (int mt = 0; mt < 2; ++mt) {
#pragma unroll
    for (int nt = 0; nt < 4; ++nt) {
      const int n    = N0 + nt * 16 + ln;
      const float bv = bias[n];
      const int mb   = M0 + mt * 16 + hi * 8;
      if constexpr (MODE == 0) {  // [B,H,T,D] bf16, scaled
        const int h = n >> 6, d = n & 63;
#pragma unroll
        for (int r = 0; r < 8; ++r) {
          const int m = mb + r, b_ = m >> 11, t = m & (T - 1);
          ((__bf16*)outp)[((size_t)(b_ * H + h) * T + t) * Dh + d] =
              f2bfh((acc[mt][nt][r] + bv) * scale);
        }
      } else if constexpr (MODE == 1) {  // [B,H,D,T] bf16, packed b128 store
        const int h = n >> 6, d = n & 63;
        const int b_ = mb >> 11, t = mb & (T - 1);
        u4 pk;
#pragma unroll
        for (int i = 0; i < 4; ++i)
          pk[i] = (unsigned)f2bf(acc[mt][nt][2 * i] + bv) |
                  ((unsigned)f2bf(acc[mt][nt][2 * i + 1] + bv) << 16);
        *(u4*)((__bf16*)outp + ((size_t)(b_ * H + h) * Dh + d) * T + t) = pk;
      } else {  // f32 row-major
#pragma unroll
        for (int r = 0; r < 8; ++r)
          ((float*)outp)[(size_t)(mb + r) * N + n] = acc[mt][nt][r] + bv;
      }
    }
  }
}

// ---------- flash attention ----------
// S^T = K * Q^T  (keys on M axis -> softmax reduce = in-lane + one shfl_xor 16)
// O^T = V^T * P^T (V^T comes for free from the transposed V projection store)
// K/V tiles staged once per block into double-buffered LDS via async DMA.
constexpr int KV   = 32;        // keys per step
constexpr int KST  = Dh + 8;    // LDS K-tile row stride (pad: bank spread)
constexpr int VST  = KV + 8;    // LDS V-tile row stride

__global__ __launch_bounds__(256) void attn_kernel(const __bf16* __restrict__ Qh,
                                                   const __bf16* __restrict__ Kh,
                                                   const __bf16* __restrict__ Vt,
                                                   __bf16* __restrict__ Y) {
  __shared__ alignas(16) __bf16 kbuf[2][KV * KST];   // 32 keys x 64 d (padded)
  __shared__ alignas(16) __bf16 vbuf[2][Dh * VST];   // 64 d x 32 keys (padded)

  const int tid  = threadIdx.x;
  const int lane = tid & 31, ln = lane & 15, hi = lane >> 4;
  const int w  = tid >> 5;
  const int bh = blockIdx.y;            // 0..63
  const int b_ = bh >> 4, h = bh & 15;
  const int q0 = blockIdx.x * 128 + w * 16;

  const __bf16* Qp = Qh + (size_t)bh * T * Dh;  // [T][D]
  const __bf16* Kp = Kh + (size_t)bh * T * Dh;  // [T][D]
  const __bf16* Vp = Vt + (size_t)bh * Dh * T;  // [D][T]  (transposed)

  // per-thread staging slots (one 16B chunk each for K and V tile)
  const int krow = tid >> 3, kcol = (tid & 7) * 8;   // 32 x 8-chunks
  const int vrow = tid >> 2, vcol = (tid & 3) * 8;   // 64 x 4-chunks

  // Q^T as WMMA-B fragments, resident for the whole kernel (D=64 -> 2 frags)
  v16bf qt[2];
#pragma unroll
  for (int kk = 0; kk < 2; ++kk)
    qt[kk] = load_b(Qp, q0 + ln, Dh, kk * 32, hi);

  // preload tile 0
  copy16(Kp + (size_t)krow * Dh + kcol, &kbuf[0][krow * KST + kcol]);
  copy16(Vp + (size_t)vrow * T + vcol, &vbuf[0][vrow * VST + vcol]);
  async_wait_all();
  __syncthreads();

  const v8f vzero = {0.f, 0.f, 0.f, 0.f, 0.f, 0.f, 0.f, 0.f};
  float m_run = -1e30f, l_run = 0.f;
  v8f o[4] = {vzero, vzero, vzero, vzero};   // O^T: 64(d) x 16(q)

  for (int jb = 0; jb < T / KV; ++jb) {
    const int cur = jb & 1, nxt = cur ^ 1;

    // kick off async DMA for the next tile into the other buffer
    if (jb + 1 < T / KV) {
      const int kb1 = (jb + 1) * KV;
      copy16(Kp + (size_t)(kb1 + krow) * Dh + kcol, &kbuf[nxt][krow * KST + kcol]);
      copy16(Vp + (size_t)vrow * T + kb1 + vcol, &vbuf[nxt][vrow * VST + vcol]);
    }

    // S^T tiles from LDS: s[mt] covers keys kb+mt*16..+15 (M) x 16 queries (N)
    v8f s[2];
#pragma unroll
    for (int mt = 0; mt < 2; ++mt) {
      v8f st = vzero;
#pragma unroll
      for (int kk = 0; kk < 2; ++kk) {
        v16bf ka = load_a_bf(&kbuf[cur][0], mt * 16 + ln, KST, kk * 32, hi);
        st = wmma_bf16(ka, qt[kk], st);
      }
      s[mt] = st;
    }

    // online softmax over keys (M axis): in-lane reduce + shfl_xor(16)
    float tm = fmaxf(s[0][0], s[1][0]);
#pragma unroll
    for (int r = 1; r < 8; ++r) tm = fmaxf(tm, fmaxf(s[0][r], s[1][r]));
    tm = fmaxf(tm, __shfl_xor(tm, 16, 32));
    const float m_new = fmaxf(m_run, tm);
    const float corr  = __builtin_amdgcn_exp2f(m_run - m_new);  // log2e in Q scale
    float rs = 0.f;
#pragma unroll
    for (int mt = 0; mt < 2; ++mt)
#pragma unroll
      for (int r = 0; r < 8; ++r) {
        float p = __builtin_amdgcn_exp2f(s[mt][r] - m_new);
        s[mt][r] = p;
        rs += p;
      }
    rs += __shfl_xor(rs, 16, 32);
    l_run = l_run * corr + rs;
    m_run = m_new;
#pragma unroll
    for (int dt = 0; dt < 4; ++dt)
#pragma unroll
      for (int r = 0; r < 8; ++r) o[dt][r] *= corr;

    // Build P^T WMMA-B fragment from S^T accumulators: pack bf16 pairs,
    // one lane-16 exchange (no LDS round trip).
    unsigned pk0[4], pk1[4], x0[4], x1[4];
#pragma unroll
    for (int i = 0; i < 4; ++i) {
      pk0[i] = (unsigned)f2bf(s[0][2 * i]) | ((unsigned)f2bf(s[0][2 * i + 1]) << 16);
      pk1[i] = (unsigned)f2bf(s[1][2 * i]) | ((unsigned)f2bf(s[1][2 * i + 1]) << 16);
    }
#pragma unroll
    for (int i = 0; i < 4; ++i) {
      x0[i] = (unsigned)__shfl_xor((int)pk0[i], 16, 32);
      x1[i] = (unsigned)__shfl_xor((int)pk1[i], 16, 32);
    }
    union { unsigned u[8]; v16bf v; } pf;
#pragma unroll
    for (int i = 0; i < 4; ++i) {
      pf.u[i]     = hi ? x1[i] : pk0[i];   // elems 0..7 : keys +0..7 / +16..23
      pf.u[4 + i] = hi ? pk1[i] : x0[i];   // elems 8..15: keys +8..15 / +24..31
    }

    // O^T += V^T(16d x 32k) * P^T(32k x 16q), 4 d-tiles, V from LDS
#pragma unroll
    for (int dt = 0; dt < 4; ++dt) {
      v16bf va = load_a_bf(&vbuf[cur][0], dt * 16 + ln, VST, 0, hi);
      o[dt] = wmma_bf16(va, pf.v, o[dt]);
    }

    // next tile's DMA must be in LDS, and all waves done reading cur
    async_wait_all();
    __syncthreads();
  }

  // epilogue: lane holds query column t=q0+ln, d = dt*16 + hi*8 + r
  const float inv = 1.0f / l_run;
  const int t = q0 + ln;
#pragma unroll
  for (int dt = 0; dt < 4; ++dt) {
    u4 pk;
#pragma unroll
    for (int i = 0; i < 4; ++i)
      pk[i] = (unsigned)f2bf(o[dt][2 * i] * inv) |
              ((unsigned)f2bf(o[dt][2 * i + 1] * inv) << 16);
    *(u4*)(Y + ((size_t)(b_ * T + t)) * C + h * Dh + dt * 16 + hi * 8) = pk;
  }
}

// ---------- host ----------
extern "C" void kernel_launch(void* const* d_in, const int* in_sizes, int n_in,
                              void* d_out, int out_size, void* d_ws, size_t ws_size,
                              hipStream_t stream) {
  const float* q  = (const float*)d_in[0];
  const float* k  = (const float*)d_in[1];
  const float* v  = (const float*)d_in[2];
  const float* Wq = (const float*)d_in[3];
  const float* bq = (const float*)d_in[4];
  const float* Wk = (const float*)d_in[5];
  const float* bk = (const float*)d_in[6];
  const float* Wv = (const float*)d_in[7];
  const float* bv = (const float*)d_in[8];
  const float* Wo = (const float*)d_in[9];
  const float* bo = (const float*)d_in[10];

  char* ws = (char*)d_ws;
  const size_t WT = (size_t)K * N * 2;     // 2 MB per transposed weight
  const size_t BF = (size_t)M * C * 2;     // 16 MB per bf16 activation buffer
  __bf16* Wqt = (__bf16*)(ws);
  __bf16* Wkt = (__bf16*)(ws + WT);
  __bf16* Wvt = (__bf16*)(ws + 2 * WT);
  __bf16* Wot = (__bf16*)(ws + 3 * WT);
  __bf16* Qh  = (__bf16*)(ws + 4 * WT);
  __bf16* Kh  = (__bf16*)(ws + 4 * WT + BF);
  __bf16* Vtp = (__bf16*)(ws + 4 * WT + 2 * BF);
  __bf16* Y   = (__bf16*)(ws + 4 * WT + 3 * BF);

  dim3 tb(256);
  dim3 gT(32, 32);
  transpose_w<<<gT, tb, 0, stream>>>(Wq, Wqt);
  transpose_w<<<gT, tb, 0, stream>>>(Wk, Wkt);
  transpose_w<<<gT, tb, 0, stream>>>(Wv, Wvt);
  transpose_w<<<gT, tb, 0, stream>>>(Wo, Wot);

  // fold softmax scale AND log2(e) into Q so attention uses native exp2
  const float qscale = 1.4426950408889634f / 8.0f;  // log2(e)/sqrt(64)

  dim3 gG(N / 128, M / 128);  // (8, 64)
  gemm_kernel<true, 0><<<gG, tb, 0, stream>>>(q, Wqt, bq, Qh, qscale);
  gemm_kernel<true, 0><<<gG, tb, 0, stream>>>(k, Wkt, bk, Kh, 1.0f);
  gemm_kernel<true, 1><<<gG, tb, 0, stream>>>(v, Wvt, bv, Vtp, 1.0f);

  dim3 gA(T / 128, Bb * H);   // (16, 64)
  attn_kernel<<<gA, tb, 0, stream>>>(Qh, Kh, Vtp, Y);

  gemm_kernel<false, 2><<<gG, tb, 0, stream>>>(Y, Wot, bo, d_out, 1.0f);
}
